// GRUCell_18872086298654
// MI455X (gfx1250) — matrix operand order repaired and checked by reading, not compiled, not run on hardware
//
#include <hip/hip_runtime.h>

typedef __attribute__((ext_vector_type(16))) _Float16 v16h;
typedef __attribute__((ext_vector_type(8)))  float    v8f;
typedef __attribute__((ext_vector_type(4)))  unsigned int u32x4;

#define HD 512
#define NMAT_HALVES (512 * 512)        // halves per weight matrix
#define XH_HALVES   (16384 * 512)      // halves for x or h
#define MT 32                          // rows per block (two 16-row m-tiles)

// ---- LDS layout (bytes), dynamic shared ----
#define A_STRIDE   1040                        // 512*2 + 16 pad
#define XT_OFF     0
#define HT_OFF     (MT * A_STRIDE)             // 33280
#define R_STRIDE   2064                        // 512*4 + 16 pad
#define ZP_OFF     (2 * MT * A_STRIDE)         // 66560
#define HP_OFF     (ZP_OFF + MT * R_STRIDE)    // 132608
#define SMEM_BYTES (HP_OFF + MT * R_STRIDE)    // 198656

// ---------------- f32 -> f16 conversion (prep) ----------------
__global__ void cvt_f32_f16(const float* __restrict__ src,
                            _Float16* __restrict__ dst, int n4) {
    int i = blockIdx.x * blockDim.x + threadIdx.x;
    if (i < n4) {
        float4 v = ((const float4*)src)[i];
        union { _Float16 h[4]; uint2 u; } o;
        o.h[0] = (_Float16)v.x; o.h[1] = (_Float16)v.y;
        o.h[2] = (_Float16)v.z; o.h[3] = (_Float16)v.w;
        ((uint2*)dst)[i] = o.u;
    }
}

__device__ __forceinline__ float gru_elem(float zp, float hp,
                                          float muz, float rz,
                                          float muh, float rh,
                                          float g0, float b0,
                                          float g2, float b2,
                                          float hold) {
    float zn = (zp - muz) * rz * g0 + b0;
    float zs = 1.0f / (1.0f + expf(-zn));
    float hn = (hp - muh) * rh * g2 + b2;
    float hh = tanf(hn);                       // torch.tan bug kept
    return zs * hold + (1.0f - zs) * hh;
}

#define WMMA(A, B, C) __builtin_amdgcn_wmma_f32_16x16x32_f16(false, A, false, B, (short)0, C, false, false)

// ---------------- fused GRU kernel ----------------
__global__ __launch_bounds__(128)
void gru_fused(const _Float16* __restrict__ x16, const _Float16* __restrict__ h16,
               const _Float16* __restrict__ w16,      // Wz,Uz,Wr,Ur,Wh,Uh packed
               const float* __restrict__ bz,  const float* __restrict__ buz,
               const float* __restrict__ br,  const float* __restrict__ bur,
               const float* __restrict__ bh,  const float* __restrict__ buh,
               const float* __restrict__ ln_g, const float* __restrict__ ln_b,
               const float* __restrict__ h_in, float* __restrict__ out) {
    extern __shared__ char smem[];
    const int tid     = threadIdx.x;
    const int lane    = tid & 31;
    const int wave    = tid >> 5;
    const int lane_lo = lane & 15;
    const int hi      = lane >> 4;           // which half of the wave
    const int row0    = blockIdx.x * MT;

    // ---- stage MT x 512 f16 tiles of x and h into LDS (async copies) ----
    {
        unsigned long long xb = (unsigned long long)x16;
        unsigned long long hb = (unsigned long long)h16;
#pragma unroll
        for (int i = 0; i < (MT * 64) / 128; ++i) {  // 16 iters: 2048 chunks/tile
            int c    = tid + i * 128;
            int r    = c >> 6;               // row 0..31
            int col  = c & 63;               // 16B chunk within row
            int goff = (row0 + r) * 1024 + col * 16;
            int lx   = XT_OFF + r * A_STRIDE + col * 16;
            int lh   = HT_OFF + r * A_STRIDE + col * 16;
            asm volatile("global_load_async_to_lds_b128 %0, %1, %2"
                         :: "v"(lx), "v"(goff), "s"(xb) : "memory");
            asm volatile("global_load_async_to_lds_b128 %0, %1, %2"
                         :: "v"(lh), "v"(goff), "s"(hb) : "memory");
        }
        asm volatile("s_wait_asynccnt 0" ::: "memory");
    }
    __syncthreads();

    union Frag { u32x4 u[2]; v16h h; };
    // A-fragment base: lane<16 reads K {0..15}, lane>=16 reads {16..31}
    const char* ax0 = smem + XT_OFF + lane_lo * A_STRIDE + hi * 32;           // m-tile 0
    const char* ah0 = smem + HT_OFF + lane_lo * A_STRIDE + hi * 32;
    const char* ax1 = ax0 + 16 * A_STRIDE;                                    // m-tile 1
    const char* ah1 = ah0 + 16 * A_STRIDE;

    for (int j = 0; j < 8; ++j) {
        const int nt = wave * 8 + j;                 // N-tile 0..31
        // B-fragment: 32 contiguous bytes of weight row (nt*16+lane_lo)
        const _Float16* wp  = w16 + ((nt * 16 + lane_lo) * 512 + hi * 16);
        const _Float16* wzp = wp;
        const _Float16* uzp = wp + 1 * NMAT_HALVES;
        const _Float16* wrp = wp + 2 * NMAT_HALVES;
        const _Float16* urp = wp + 3 * NMAT_HALVES;
        const _Float16* whp = wp + 4 * NMAT_HALVES;
        const _Float16* uhp = wp + 5 * NMAT_HALVES;

        v8f az0 = {}, ar0 = {}, a10 = {}, a20 = {};
        v8f az1 = {}, ar1 = {}, a11 = {}, a21 = {};
        for (int ks = 0; ks < 16; ++ks) {            // K = 512 = 16 x 32
            int kb = ks * 64;                        // bytes into A row
            Frag fx0, fh0, fx1, fh1;
            fx0.u[0] = *(const u32x4*)(ax0 + kb);
            fx0.u[1] = *(const u32x4*)(ax0 + kb + 16);
            fh0.u[0] = *(const u32x4*)(ah0 + kb);
            fh0.u[1] = *(const u32x4*)(ah0 + kb + 16);
            fx1.u[0] = *(const u32x4*)(ax1 + kb);
            fx1.u[1] = *(const u32x4*)(ax1 + kb + 16);
            fh1.u[0] = *(const u32x4*)(ah1 + kb);
            fh1.u[1] = *(const u32x4*)(ah1 + kb + 16);
            int kw = ks * 32;                        // halves into W row
            v16h wz_ = *(const v16h*)(wzp + kw);
            v16h uz_ = *(const v16h*)(uzp + kw);
            v16h wr_ = *(const v16h*)(wrp + kw);
            v16h ur_ = *(const v16h*)(urp + kw);
            v16h wh_ = *(const v16h*)(whp + kw);
            v16h uh_ = *(const v16h*)(uhp + kw);
            // 12 WMMAs against 6 weight-fragment loads (two m-tiles share B)
            az0 = WMMA(fx0.h, wz_, az0);  az1 = WMMA(fx1.h, wz_, az1);
            az0 = WMMA(fh0.h, uz_, az0);  az1 = WMMA(fh1.h, uz_, az1);
            ar0 = WMMA(fx0.h, wr_, ar0);  ar1 = WMMA(fx1.h, wr_, ar1);
            ar0 = WMMA(fh0.h, ur_, ar0);  ar1 = WMMA(fh1.h, ur_, ar1);
            a10 = WMMA(fx0.h, wh_, a10);  a11 = WMMA(fx1.h, wh_, a11);
            a20 = WMMA(fh0.h, uh_, a20);  a21 = WMMA(fh1.h, uh_, a21);
        }

        // fuse biases + r-gating, spill pre-activations to LDS (f32)
        const int n   = nt * 16 + lane_lo;
        const float bz_s = bz[n] + buz[n];
        const float br_s = br[n] + bur[n];
        const float bh_s = bh[n];
        const float bu_s = buh[n];
        char* zdst = smem + ZP_OFF + (hi * 8) * R_STRIDE + n * 4;
        char* hdst = smem + HP_OFF + (hi * 8) * R_STRIDE + n * 4;
#pragma unroll
        for (int i = 0; i < 8; ++i) {               // VGPR i -> row i (+8 if hi)
            float zp0 = az0[i] + bz_s;
            float rp0 = ar0[i] + br_s;
            float hp0 = a10[i] + bh_s + rp0 * (a20[i] + bu_s);
            *(float*)(zdst + i * R_STRIDE) = zp0;
            *(float*)(hdst + i * R_STRIDE) = hp0;
            float zp1 = az1[i] + bz_s;
            float rp1 = ar1[i] + br_s;
            float hp1 = a11[i] + bh_s + rp1 * (a21[i] + bu_s);
            *(float*)(zdst + (16 + i) * R_STRIDE) = zp1;
            *(float*)(hdst + (16 + i) * R_STRIDE) = hp1;
        }
    }
    __syncthreads();

    // ---- layernorm stats: 4 threads per row (32 rows), 128 cols each ----
    const int row = tid >> 2;
    const int p   = tid & 3;
    const char* zrow = smem + ZP_OFF + row * R_STRIDE;
    const char* hrow = smem + HP_OFF + row * R_STRIDE;
    float sz = 0.f, szz = 0.f, sh = 0.f, shh = 0.f;
#pragma unroll
    for (int c = 0; c < 32; ++c) {
        float4 zv = *(const float4*)(zrow + (p * 128 + c * 4) * 4);
        float4 hv = *(const float4*)(hrow + (p * 128 + c * 4) * 4);
        sz  += zv.x + zv.y + zv.z + zv.w;
        szz += zv.x * zv.x + zv.y * zv.y + zv.z * zv.z + zv.w * zv.w;
        sh  += hv.x + hv.y + hv.z + hv.w;
        shh += hv.x * hv.x + hv.y * hv.y + hv.z * hv.z + hv.w * hv.w;
    }
#pragma unroll
    for (int m = 1; m < 4; m <<= 1) {               // butterfly within 4-lane group
        sz  += __shfl_xor(sz,  m, 4);
        szz += __shfl_xor(szz, m, 4);
        sh  += __shfl_xor(sh,  m, 4);
        shh += __shfl_xor(shh, m, 4);
    }
    const float inv = 1.0f / 512.0f;
    float muz = sz * inv, muh = sh * inv;
    float rz = rsqrtf(szz * inv - muz * muz + 1e-5f);
    float rh = rsqrtf(shh * inv - muh * muh + 1e-5f);

    // ---- normalize + sigmoid/tan + gate + store ----
    const float* g0 = ln_g;          const float* b0 = ln_b;
    const float* g2 = ln_g + 1024;   const float* b2 = ln_b + 1024;
    const float* hg = h_in + (size_t)(row0 + row) * HD;
    float* og       = out  + (size_t)(row0 + row) * HD;
#pragma unroll 4
    for (int c = 0; c < 32; ++c) {
        int nn = p * 128 + c * 4;
        float4 zv = *(const float4*)(zrow + nn * 4);
        float4 hv = *(const float4*)(hrow + nn * 4);
        float4 gz = *(const float4*)(g0 + nn);
        float4 bzv = *(const float4*)(b0 + nn);
        float4 gh = *(const float4*)(g2 + nn);
        float4 bhv = *(const float4*)(b2 + nn);
        float4 ho = *(const float4*)(hg + nn);
        float4 o;
        o.x = gru_elem(zv.x, hv.x, muz, rz, muh, rh, gz.x, bzv.x, gh.x, bhv.x, ho.x);
        o.y = gru_elem(zv.y, hv.y, muz, rz, muh, rh, gz.y, bzv.y, gh.y, bhv.y, ho.y);
        o.z = gru_elem(zv.z, hv.z, muz, rz, muh, rh, gz.z, bzv.z, gh.z, bhv.z, ho.z);
        o.w = gru_elem(zv.w, hv.w, muz, rz, muh, rh, gz.w, bzv.w, gh.w, bhv.w, ho.w);
        *(float4*)(og + nn) = o;
    }
}

extern "C" void kernel_launch(void* const* d_in, const int* in_sizes, int n_in,
                              void* d_out, int out_size, void* d_ws, size_t ws_size,
                              hipStream_t stream) {
    const float* x   = (const float*)d_in[0];
    const float* h   = (const float*)d_in[1];
    const float* Wz  = (const float*)d_in[2];
    const float* bz  = (const float*)d_in[3];
    const float* Uz  = (const float*)d_in[4];
    const float* buz = (const float*)d_in[5];
    const float* Wr  = (const float*)d_in[6];
    const float* br  = (const float*)d_in[7];
    const float* Ur  = (const float*)d_in[8];
    const float* bur = (const float*)d_in[9];
    const float* Wh  = (const float*)d_in[10];
    const float* bh  = (const float*)d_in[11];
    const float* Uh  = (const float*)d_in[12];
    const float* buh = (const float*)d_in[13];
    const float* lng = (const float*)d_in[14];
    const float* lnb = (const float*)d_in[15];
    float* out = (float*)d_out;

    _Float16* ws16 = (_Float16*)d_ws;
    _Float16* x16 = ws16;
    _Float16* h16 = ws16 + XH_HALVES;
    _Float16* w16 = ws16 + 2 * (size_t)XH_HALVES;   // 6 matrices follow

    // prep: f32 -> f16 (x, h, 6 weights)
    cvt_f32_f16<<<XH_HALVES / 4 / 256, 256, 0, stream>>>(x, x16, XH_HALVES / 4);
    cvt_f32_f16<<<XH_HALVES / 4 / 256, 256, 0, stream>>>(h, h16, XH_HALVES / 4);
    const float* wsrc[6] = {Wz, Uz, Wr, Ur, Wh, Uh};
    for (int i = 0; i < 6; ++i)
        cvt_f32_f16<<<NMAT_HALVES / 4 / 256, 256, 0, stream>>>(
            wsrc[i], w16 + (size_t)i * NMAT_HALVES, NMAT_HALVES / 4);

    // fused GEMM + layernorm + GRU gate: 512 blocks x 128 threads (4 waves)
    gru_fused<<<16384 / MT, 128, SMEM_BYTES, stream>>>(
        x16, h16, w16, bz, buz, br, bur, bh, buh, lng, lnb, h, out);
}
